// AttHierarchicalGround_18545668784795
// MI455X (gfx1250) — matrix operand order repaired and checked by reading, not compiled, not run on hardware
//
#include <hip/hip_runtime.h>
#include <hip/hip_bf16.h>
#include <math.h>

// ---------------- problem constants ----------------
namespace {
constexpr int cB = 8, cT = 120, cNB = 40, cVD = 2053, cWD = 300;
constexpr int cH = 512, cE = 256, cNH = 8, cHD = 64, cNSEG = 10, cSEG = 12;
constexpr int cM = cB * cT * cNB;   // 38400 rows of ve / Y
}

typedef __bf16 v16bf __attribute__((ext_vector_type(16)));
typedef float  v8f   __attribute__((ext_vector_type(8)));

__device__ inline float sigm(float x) { return 1.f / (1.f + expf(-x)); }

// ======================================================================
// Weight pre-swizzle: fp32 (K x N, row-major) -> bf16 WMMA B-fragment
// layout Wsw[kt][nt][lane][16].  Lane L covers column n = nt*16 + (L&15);
// element i covers k = kt*32 + ((L&16)?8:0) + i + (i>=8 ? 8 : 0).
// ======================================================================
__global__ void swizzle_w(const float* __restrict__ W, __bf16* __restrict__ Wsw,
                          int K, int N) {
    int tid  = blockIdx.x * blockDim.x + threadIdx.x;
    int lane = tid & 31;
    int rest = tid >> 5;
    int nTiles = N >> 4;
    int kTiles = K >> 5;
    int nt = rest % nTiles;
    int kt = rest / nTiles;
    if (kt >= kTiles) return;
    int n    = nt * 16 + (lane & 15);
    int kOff = kt * 32 + ((lane & 16) ? 8 : 0);
    __bf16* dst = Wsw + ((size_t)(kt * nTiles + nt) * 32 + lane) * 16;
#pragma unroll
    for (int i = 0; i < 16; i++) dst[i] = (__bf16)W[(size_t)(kOff + i + (i >= 8 ? 8 : 0)) * N + n];
}

// ======================================================================
// Async copy of one k-step of B fragments (16 tiles = 16KB) into LDS.
// Wave w copies tiles 4w..4w+3; each lane moves its 32B fragment as
// 2x global_load_async_to_lds_b128 (ASYNCcnt, in-order completion).
// ======================================================================
__device__ inline void async_fill_b(const __bf16* __restrict__ wbase,
                                    __bf16* dst_base, int waveId, int lane) {
#pragma unroll
    for (int jj = 0; jj < 4; jj++) {
        int j = waveId * 4 + jj;
        const __bf16* src = wbase + (size_t)j * 512 + lane * 16;
        unsigned ldsOff = (unsigned)(size_t)(dst_base + (size_t)j * 512 + lane * 16);
        asm volatile("global_load_async_to_lds_b128 %0, %1, off"
                     :: "v"(ldsOff), "v"(src) : "memory");
        asm volatile("global_load_async_to_lds_b128 %0, %1, off offset:16"
                     :: "v"(ldsOff), "v"(src) : "memory");
    }
}

// ======================================================================
// ve kernel: ve = relu(videos[:,:2048] @ W_ev + b_ev) + relu(videos[:,2048:2053] @ W_el + b_el)
// One wave -> 16 rows x full 256 cols (videos read exactly once from HBM).
// B fragments are async-staged into LDS (double buffered) and shared by the
// 4 waves of the block: cuts L2 weight traffic 4x (2.4GB -> 600MB).
// Grid is exactly 2400 waves (600 x 128), so no wave skips the barriers.
// ======================================================================
__global__ void __launch_bounds__(128, 1)
ve_kernel(const float* __restrict__ videos, const __bf16* __restrict__ Wsw,
          const float* __restrict__ b_ev, const float* __restrict__ W_el,
          const float* __restrict__ b_el, float* __restrict__ ve) {
    __shared__ __align__(32) __bf16 lbuf[2][16 * 512];   // 2 x 16KB
    int wave   = (blockIdx.x * blockDim.x + threadIdx.x) >> 5;
    int waveId = (threadIdx.x >> 5) & 3;
    int lane   = threadIdx.x & 31;
    int mBase  = wave * 16;
    int row    = mBase + (lane & 15);
    const float* arow = videos + (size_t)row * cVD + ((lane & 16) ? 8 : 0);

    v8f acc[16] = {};
    async_fill_b(Wsw, &lbuf[0][0], waveId, lane);        // prologue fill (cnt=8)
    for (int kt = 0; kt < 64; kt++) {
        int cur = kt & 1;
        if (kt + 1 < 64) {
            async_fill_b(Wsw + (size_t)(kt + 1) * 16 * 512, &lbuf[cur ^ 1][0],
                         waveId, lane);                  // cnt <= 16
            asm volatile("s_wait_asynccnt 0x8" ::: "memory");  // oldest 8 (cur) done
        } else {
            asm volatile("s_wait_asynccnt 0x0" ::: "memory");
        }
        __syncthreads();                                 // buf[cur] visible to all

        int k = kt * 32;
        if (k + 32 < 2048) __builtin_prefetch(arow + k + 32, 0, 3);
        v16bf a;
        const float* ap = arow + k;
#pragma unroll
        for (int i = 0; i < 8; i++) a[i] = (__bf16)ap[i];
#pragma unroll
        for (int i = 0; i < 8; i++) a[8 + i] = (__bf16)ap[16 + i];
#pragma unroll
        for (int j = 0; j < 16; j++) {
            v16bf bfrag = *(const v16bf*)&lbuf[cur][(size_t)j * 512 + lane * 16];
            acc[j] = __builtin_amdgcn_wmma_f32_16x16x32_bf16(
                false, a, false, bfrag, (short)0, acc[j], false, false);
        }
        __syncthreads();                                 // all reads of cur done
    }

    int mloc = (lane & 16) ? 8 : 0;
    float v5[8][5];
#pragma unroll
    for (int r = 0; r < 8; r++) {
        int m = mBase + r + mloc;
#pragma unroll
        for (int i = 0; i < 5; i++) v5[r][i] = videos[(size_t)m * cVD + 2048 + i];
    }
    for (int j = 0; j < 16; j++) {
        int n = j * 16 + (lane & 15);
        float bev = b_ev[n], bel = b_el[n];
        float wl[5];
#pragma unroll
        for (int i = 0; i < 5; i++) wl[i] = W_el[i * cE + n];
#pragma unroll
        for (int r = 0; r < 8; r++) {
            int m = mBase + r + mloc;
            float el = bel;
#pragma unroll
            for (int i = 0; i < 5; i++) el += v5[r][i] * wl[i];
            ve[(size_t)m * cE + n] = fmaxf(acc[j][r] + bev, 0.f) + fmaxf(el, 0.f);
        }
    }
}

// ======================================================================
// Generic WMMA GEMM: C[m,n] = act( A(fp32) @ Wsw(bf16) + bias[n] + addend[m,n] )
// One wave -> 16 rows x 128 cols.  act: 0=none, 1=relu.
// ======================================================================
__global__ void __launch_bounds__(128, 1)
gemm_wmma(const float* __restrict__ A, const __bf16* __restrict__ Wsw,
          const float* __restrict__ bias, const float* __restrict__ addend,
          float* __restrict__ C, int M, int K, int N, int lda, int ldc,
          int act) {
    int wave = (blockIdx.x * blockDim.x + threadIdx.x) >> 5;
    int lane = threadIdx.x & 31;
    int nGroups = N >> 7;
    int mTiles  = (M + 15) >> 4;
    if (wave >= mTiles * nGroups) return;
    int mTile = wave / nGroups;
    int ng    = wave % nGroups;
    int row   = mTile * 16 + (lane & 15);
    if (row >= M) row = M - 1;
    const float* arow = A + (size_t)row * lda + ((lane & 16) ? 8 : 0);
    int nTiles = N >> 4;

    v8f acc[8] = {};
    for (int k = 0; k < K; k += 32) {
        if (k + 32 < K) __builtin_prefetch(arow + k + 32, 0, 3);
        v16bf a;
        const float* ap = arow + k;
#pragma unroll
        for (int i = 0; i < 8; i++) a[i] = (__bf16)ap[i];
#pragma unroll
        for (int i = 0; i < 8; i++) a[8 + i] = (__bf16)ap[16 + i];
        const __bf16* wbase = Wsw + ((size_t)(k >> 5) * nTiles + ng * 8) * 512;
#pragma unroll
        for (int j = 0; j < 8; j++) {
            v16bf bfrag = *(const v16bf*)(wbase + (size_t)j * 512 + lane * 16);
            acc[j] = __builtin_amdgcn_wmma_f32_16x16x32_bf16(
                false, a, false, bfrag, (short)0, acc[j], false, false);
        }
    }

    int mloc = (lane & 16) ? 8 : 0;
#pragma unroll
    for (int j = 0; j < 8; j++) {
        int n = ng * 128 + j * 16 + (lane & 15);
        float bi = bias ? bias[n] : 0.f;
#pragma unroll
        for (int r = 0; r < 8; r++) {
            int m = mTile * 16 + r + mloc;
            if (m < M) {
                float v = acc[j][r] + bi;
                if (addend) v += addend[(size_t)m * ldc + n];
                if (act) v = fmaxf(v, 0.f);
                C[(size_t)m * ldc + n] = v;
            }
        }
    }
}

// ======================================================================
// Small dense helpers (VALU): out[b,n] = act( in[b,:] @ W[rowOff:rowOff+K, :] + bias )
// ======================================================================
__global__ void vecmat(const float* __restrict__ in, const float* __restrict__ W,
                       const float* __restrict__ bias, float* __restrict__ out,
                       int Bn, int K, int N, int rowOff, int act) {
    int idx = blockIdx.x * blockDim.x + threadIdx.x;
    if (idx >= Bn * N) return;
    int b = idx / N, n = idx % N;
    float s = bias ? bias[n] : 0.f;
    for (int j = 0; j < K; j++) s += in[b * K + j] * W[(size_t)(j + rowOff) * N + n];
    if (act) s = fmaxf(s, 0.f);
    out[idx] = s;
}

__global__ void vecmat2(const float* __restrict__ in1, const float* __restrict__ in2,
                        const float* __restrict__ W, const float* __restrict__ bias,
                        float* __restrict__ out, int Bn, int K1, int K2, int N, int act) {
    int idx = blockIdx.x * blockDim.x + threadIdx.x;
    if (idx >= Bn * N) return;
    int b = idx / N, n = idx % N;
    float s = bias ? bias[n] : 0.f;
    for (int j = 0; j < K1; j++) s += in1[b * K1 + j] * W[(size_t)j * N + n];
    for (int j = 0; j < K2; j++) s += in2[b * K2 + j] * W[(size_t)(K1 + j) * N + n];
    if (act) s = fmaxf(s, 0.f);
    out[idx] = s;
}

__global__ void zero_f(float* p, int n) {
    int i = blockIdx.x * blockDim.x + threadIdx.x;
    if (i < n) p[i] = 0.f;
}

// ======================================================================
// attend scores: s_{sub,obj}[row] = sum_e tanh(Y[row,e] + wt[b,e]) * w_sp2[e]
// ======================================================================
__global__ void attend_scores(const float* __restrict__ Y, const float* __restrict__ wts,
                              const float* __restrict__ wto, const float* __restrict__ w_sp2,
                              float* __restrict__ s_sub, float* __restrict__ s_obj) {
    int row = blockIdx.x;               // 0..38399
    int b   = row / (cT * cNB);
    int e   = threadIdx.x;              // 0..255
    float y  = Y[(size_t)row * cE + e];
    float w2 = w_sp2[e];
    __shared__ float rs[256], ro[256];
    rs[e] = tanhf(y + wts[b * cE + e]) * w2;
    ro[e] = tanhf(y + wto[b * cE + e]) * w2;
    __syncthreads();
    for (int st = 128; st > 0; st >>= 1) {
        if (e < st) { rs[e] += rs[e + st]; ro[e] += ro[e + st]; }
        __syncthreads();
    }
    if (e == 0) { s_sub[row] = rs[0]; s_obj[row] = ro[0]; }
}

__global__ void softmax_nb(const float* __restrict__ s_sub, const float* __restrict__ s_obj,
                           float* __restrict__ a_sub, float* __restrict__ a_obj) {
    int i = blockIdx.x * blockDim.x + threadIdx.x;
    if (i >= cB * cT) return;
    for (int w = 0; w < 2; w++) {
        const float* s = w ? s_obj : s_sub;
        float* a = w ? a_obj : a_sub;
        float mx = -1e30f;
        for (int n = 0; n < cNB; n++) mx = fmaxf(mx, s[i * cNB + n]);
        float sum = 0.f;
        for (int n = 0; n < cNB; n++) sum += expf(s[i * cNB + n] - mx);
        float inv = 1.f / sum;
        for (int n = 0; n < cNB; n++) a[i * cNB + n] = expf(s[i * cNB + n] - mx) * inv;
    }
}

// ori_x[bt, c<256]  = sum_n a_sub*ve[..,c]    + relu(sum_n a_obj*W_o2s[n,c] + b_o2s[c])
// ori_x[bt, 256+e]  = sum_n a_obj*ve[..,e]    + relu(sum_n a_sub*W_s2o[n,e] + b_s2o[e])
__global__ void build_orix(const float* __restrict__ ve, const float* __restrict__ a_sub,
                           const float* __restrict__ a_obj, const float* __restrict__ W_s2o,
                           const float* __restrict__ b_s2o, const float* __restrict__ W_o2s,
                           const float* __restrict__ b_o2s, float* __restrict__ orix) {
    int idx = blockIdx.x * blockDim.x + threadIdx.x;
    if (idx >= cB * cT * cH) return;
    int bt = idx >> 9;
    int c  = idx & 511;
    const float* as = a_sub + bt * cNB;
    const float* ao = a_obj + bt * cNB;
    float feat = 0.f, cross = 0.f;
    if (c < cE) {
        for (int n = 0; n < cNB; n++) {
            feat  += as[n] * ve[((size_t)bt * cNB + n) * cE + c];
            cross += ao[n] * W_o2s[n * cE + c];
        }
        orix[idx] = feat + fmaxf(cross + b_o2s[c], 0.f);
    } else {
        int e = c - cE;
        for (int n = 0; n < cNB; n++) {
            feat  += ao[n] * ve[((size_t)bt * cNB + n) * cE + e];
            cross += as[n] * W_s2o[n * cE + e];
        }
        orix[idx] = feat + fmaxf(cross + b_s2o[e], 0.f);
    }
}

// ======================================================================
// transformer attention (NH=8, HD=64, T=120)
// ======================================================================
__global__ void attn_scores(const float* __restrict__ q, const float* __restrict__ kk,
                            float* __restrict__ att) {
    int bid = blockIdx.x;               // b*NH*T
    int qt = bid % cT;
    int h  = (bid / cT) % cNH;
    int b  = bid / (cT * cNH);
    int tid = threadIdx.x;
    __shared__ float sc[cT];
    __shared__ float red[2];
    const float* qp = q + ((size_t)(b * cT + qt)) * cH + h * cHD;
    if (tid < cT) {
        const float* kp = kk + ((size_t)(b * cT + tid)) * cH + h * cHD;
        float s = 0.f;
        for (int d = 0; d < cHD; d++) s += qp[d] * kp[d];
        sc[tid] = s * 0.125f;           // 1/sqrt(64)
    }
    __syncthreads();
    if (tid == 0) {
        float mx = sc[0];
        for (int i = 1; i < cT; i++) mx = fmaxf(mx, sc[i]);
        float sum = 0.f;
        for (int i = 0; i < cT; i++) sum += expf(sc[i] - mx);
        red[0] = mx; red[1] = sum;
    }
    __syncthreads();
    if (tid < cT)
        att[((size_t)((b * cNH + h) * cT + qt)) * cT + tid] = expf(sc[tid] - red[0]) / red[1];
}

__global__ void attn_out(const float* __restrict__ att, const float* __restrict__ v,
                         float* __restrict__ ao) {
    int idx = blockIdx.x * blockDim.x + threadIdx.x;
    if (idx >= cB * cT * cH) return;
    int d  = idx % cHD;
    int h  = (idx / cHD) % cNH;
    int qt = (idx / cH) % cT;
    int b  = idx / (cT * cH);
    const float* ar = att + ((size_t)((b * cNH + h) * cT + qt)) * cT;
    float s = 0.f;
    for (int kt = 0; kt < cT; kt++)
        s += ar[kt] * v[((size_t)(b * cT + kt)) * cH + h * cHD + d];
    ao[idx] = s;
}

__global__ void layernorm_k(const float* __restrict__ in, const float* __restrict__ g,
                            const float* __restrict__ bb, float* __restrict__ out) {
    int row = blockIdx.x;
    int tid = threadIdx.x;              // 256 threads, 512 cols
    const float* r = in + (size_t)row * cH;
    float x0 = r[tid], x1 = r[tid + 256];
    __shared__ float red[256];
    red[tid] = x0 + x1; __syncthreads();
    for (int s = 128; s > 0; s >>= 1) { if (tid < s) red[tid] += red[tid + s]; __syncthreads(); }
    float mean = red[0] * (1.f / 512.f);
    __syncthreads();
    float d0 = x0 - mean, d1 = x1 - mean;
    red[tid] = d0 * d0 + d1 * d1; __syncthreads();
    for (int s = 128; s > 0; s >>= 1) { if (tid < s) red[tid] += red[tid + s]; __syncthreads(); }
    float inv = rsqrtf(red[0] * (1.f / 512.f) + 1e-5f);
    out[(size_t)row * cH + tid]       = d0 * inv * g[tid] + bb[tid];
    out[(size_t)row * cH + tid + 256] = d1 * inv * g[tid + 256] + bb[tid + 256];
}

// ======================================================================
// segment gather, beta2, LSTM, beta1, final output
// ======================================================================
__global__ void seg_gather(const float* __restrict__ within, float* __restrict__ seg) {
    int idx = blockIdx.x * blockDim.x + threadIdx.x;
    if (idx >= cB * cNSEG * cH) return;
    int h = idx & 511;
    int s = (idx >> 9) % cNSEG;
    int b = idx / (cNSEG * cH);
    int t = 11 + 12 * s;                // IDX = [11,23,...,119]
    seg[idx] = within[((size_t)(b * cT + t)) * cH + h];
}

__global__ void beta2_scores(const float* __restrict__ seg, const float* __restrict__ W_tb1,
                             const float* __restrict__ trterm, const float* __restrict__ w_tb2,
                             float* __restrict__ score2) {
    int bs = blockIdx.x;                // b*10+s
    int b  = bs / cNSEG;
    int e  = threadIdx.x;               // 512
    const float* row = seg + (size_t)bs * cH;
    float dot = trterm[b * cH + e];
    for (int j = 0; j < cH; j++) dot += row[j] * W_tb1[(size_t)j * cH + e];
    __shared__ float red[512];
    red[e] = tanhf(dot) * w_tb2[e];
    __syncthreads();
    for (int st = 256; st > 0; st >>= 1) { if (e < st) red[e] += red[e + st]; __syncthreads(); }
    if (e == 0) score2[bs] = red[0];
}

__global__ void softmax10(const float* __restrict__ sc, float* __restrict__ beta2) {
    int b = threadIdx.x;
    if (b >= cB) return;
    float mx = -1e30f;
    for (int s = 0; s < cNSEG; s++) mx = fmaxf(mx, sc[b * cNSEG + s]);
    float sum = 0.f;
    for (int s = 0; s < cNSEG; s++) sum += expf(sc[b * cNSEG + s] - mx);
    for (int s = 0; s < cNSEG; s++) beta2[b * cNSEG + s] = expf(sc[b * cNSEG + s] - mx) / sum;
}

__global__ void lstm_gates(const float* __restrict__ seg, const float* __restrict__ h,
                           const float* __restrict__ W_ih, const float* __restrict__ W_hh,
                           const float* __restrict__ b_ih, const float* __restrict__ b_hh,
                           float* __restrict__ gates, int t) {
    int idx = blockIdx.x * blockDim.x + threadIdx.x;
    if (idx >= cB * 4 * cH) return;
    int b = idx >> 11;
    int u = idx & 2047;
    const float* x  = seg + ((size_t)b * cNSEG + t) * cH;
    const float* hb = h + b * cH;
    float g = b_ih[u] + b_hh[u];
    for (int j = 0; j < cH; j++)
        g += x[j] * W_ih[(size_t)u * cH + j] + hb[j] * W_hh[(size_t)u * cH + j];
    gates[idx] = g;
}

__global__ void lstm_update(const float* __restrict__ gates, float* __restrict__ h,
                            float* __restrict__ c, float* __restrict__ hT_out,
                            float* __restrict__ cT_out, int last) {
    int idx = blockIdx.x * blockDim.x + threadIdx.x;
    if (idx >= cB * cH) return;
    int b = idx >> 9;
    int j = idx & 511;
    float i_ = gates[b * 2048 + j];
    float f_ = gates[b * 2048 + 512 + j];
    float g_ = gates[b * 2048 + 1024 + j];
    float o_ = gates[b * 2048 + 1536 + j];
    float cc = sigm(f_) * c[idx] + sigm(i_) * tanhf(g_);
    float hh = sigm(o_) * tanhf(cc);
    c[idx] = cc; h[idx] = hh;
    if (last) { hT_out[idx] = hh; cT_out[idx] = cc; }
}

__global__ void beta1_scores(const float* __restrict__ Ylw, const float* __restrict__ hvec,
                             const float* __restrict__ w_l2, float* __restrict__ s1) {
    int bt = blockIdx.x;                // 0..959
    int b  = bt / cT;
    int e  = threadIdx.x;               // 512
    float dot = Ylw[(size_t)bt * cH + e] + hvec[b * cH + e];
    __shared__ float red[512];
    red[e] = tanhf(dot) * w_l2[e];
    __syncthreads();
    for (int st = 256; st > 0; st >>= 1) { if (e < st) red[e] += red[e + st]; __syncthreads(); }
    if (e == 0) s1[bt] = red[0];
}

__global__ void softmax_t(const float* __restrict__ s1, float* __restrict__ beta1) {
    int b = blockIdx.x;
    int t = threadIdx.x;
    __shared__ float sc[cT];
    __shared__ float red[2];
    if (t < cT) sc[t] = s1[b * cT + t];
    __syncthreads();
    if (t == 0) {
        float mx = sc[0];
        for (int i = 1; i < cT; i++) mx = fmaxf(mx, sc[i]);
        float sum = 0.f;
        for (int i = 0; i < cT; i++) sum += expf(sc[i] - mx);
        red[0] = mx; red[1] = sum;
    }
    __syncthreads();
    if (t < cT) beta1[b * cT + t] = expf(sc[t] - red[0]) / red[1];
}

__global__ void final_out(const float* __restrict__ within, const float* __restrict__ beta1,
                          const float* __restrict__ beta2, float* __restrict__ out) {
    int idx = blockIdx.x * blockDim.x + threadIdx.x;
    if (idx >= cB * cH) return;
    int b = idx >> 9;
    int h = idx & 511;
    float s = 0.f;
    for (int t = 0; t < cT; t++) {
        float w = beta1[b * cT + t] + beta2[b * cNSEG + t / cSEG];
        s += w * within[((size_t)(b * cT + t)) * cH + h];
    }
    out[idx] = s;
}

// ======================================================================
// host-side orchestration
// ======================================================================
extern "C" void kernel_launch(void* const* d_in, const int* in_sizes, int n_in,
                              void* d_out, int out_size, void* d_ws, size_t ws_size,
                              hipStream_t stream) {
    (void)in_sizes; (void)n_in; (void)out_size; (void)ws_size;
    const float* videos    = (const float*)d_in[0];
    const float* sub_glove = (const float*)d_in[1];
    const float* obj_glove = (const float*)d_in[2];
    const float* W_ev  = (const float*)d_in[3];
    const float* b_ev  = (const float*)d_in[4];
    const float* W_el  = (const float*)d_in[5];
    const float* b_el  = (const float*)d_in[6];
    const float* W_ew  = (const float*)d_in[7];
    const float* b_ew  = (const float*)d_in[8];
    const float* W_sp1 = (const float*)d_in[9];
    const float* b_sp1 = (const float*)d_in[10];
    const float* w_sp2 = (const float*)d_in[11];
    const float* W_s2o = (const float*)d_in[12];
    const float* b_s2o = (const float*)d_in[13];
    const float* W_o2s = (const float*)d_in[14];
    const float* b_o2s = (const float*)d_in[15];
    const float* W_tv  = (const float*)d_in[16];
    const float* b_tv  = (const float*)d_in[17];
    const float* Wq    = (const float*)d_in[18];
    const float* bq    = (const float*)d_in[19];
    const float* Wk    = (const float*)d_in[20];
    const float* bk    = (const float*)d_in[21];
    const float* Wv    = (const float*)d_in[22];
    const float* bv    = (const float*)d_in[23];
    const float* Wo    = (const float*)d_in[24];
    const float* bo    = (const float*)d_in[25];
    const float* g1    = (const float*)d_in[26];
    const float* be1   = (const float*)d_in[27];
    const float* Wf1   = (const float*)d_in[28];
    const float* bf1   = (const float*)d_in[29];
    const float* Wf2   = (const float*)d_in[30];
    const float* bf2   = (const float*)d_in[31];
    const float* g2    = (const float*)d_in[32];
    const float* be2   = (const float*)d_in[33];
    const float* W_tr  = (const float*)d_in[34];
    const float* b_tr  = (const float*)d_in[35];
    const float* W_tb1 = (const float*)d_in[36];
    const float* b_tb1 = (const float*)d_in[37];
    const float* w_tb2 = (const float*)d_in[38];
    const float* W_ih  = (const float*)d_in[39];
    const float* W_hh  = (const float*)d_in[40];
    const float* b_ih  = (const float*)d_in[41];
    const float* b_hh  = (const float*)d_in[42];
    const float* W_l1  = (const float*)d_in[43];
    const float* b_l1  = (const float*)d_in[44];
    const float* w_l2  = (const float*)d_in[45];

    // ---- output regions (flat, return order) ----
    float* out        = (float*)d_out;
    float* out_output = out;            // 8*512
    float* out_hT     = out + 4096;     // 8*512
    float* out_cT     = out + 8192;     // 8*512
    float* out_sub    = out + 12288;    // 8*120*40
    float* out_obj    = out + 50688;    // 8*120*40
    float* out_b1     = out + 89088;    // 8*120
    float* out_b2     = out + 90048;    // 8*10

    // ---- workspace layout (bytes) ----
    char* ws = (char*)d_ws;
    __bf16* wev_sw  = (__bf16*)(ws + 0);
    __bf16* w1a_sw  = (__bf16*)(ws + 1048576);
    __bf16* wtv_sw  = (__bf16*)(ws + 1179648);
    __bf16* wq_sw   = (__bf16*)(ws + 1703936);
    __bf16* wk_sw   = (__bf16*)(ws + 2228224);
    __bf16* wv_sw   = (__bf16*)(ws + 2752512);
    __bf16* wo_sw   = (__bf16*)(ws + 3276800);
    __bf16* wf1_sw  = (__bf16*)(ws + 3801088);
    __bf16* wf2_sw  = (__bf16*)(ws + 4325376);
    __bf16* wl1a_sw = (__bf16*)(ws + 4849664);
    float* sub_e  = (float*)(ws + 5373952);
    float* obj_e  = (float*)(ws + 5382144);
    float* wts    = (float*)(ws + 5390336);
    float* wto    = (float*)(ws + 5398528);
    float* s_sub  = (float*)(ws + 5406720);
    float* s_obj  = (float*)(ws + 5560320);
    float* trbuf  = (float*)(ws + 5713920);
    float* trterm = (float*)(ws + 5730304);
    float* hvec   = (float*)(ws + 5746688);
    float* sc2    = (float*)(ws + 5763072);
    float* gates  = (float*)(ws + 5763584);
    float* hbuf   = (float*)(ws + 5829120);
    float* cbuf   = (float*)(ws + 5845504);
    float* s1     = (float*)(ws + 5861888);
    float* seg    = (float*)(ws + 5865984);
    const size_t OFF_VE = 6291456;
    const size_t OFF_Y  = 45613056;
    float* ve   = (float*)(ws + OFF_VE);      // 39321600 B, dead after build_orix
    float* Y    = (float*)(ws + OFF_Y);       // 39321600 B, dead after attend_scores
    // aliases into dead regions (lifetimes verified):
    float* orix   = (float*)(ws + OFF_Y);               // overlays Y (dead)
    float* Ylw    = (float*)(ws + OFF_Y + 1966080);
    float* xbuf   = (float*)(ws + OFF_VE + 0);          // overlays ve (dead)
    float* qbuf   = (float*)(ws + OFF_VE + 1966080);
    float* kbuf   = (float*)(ws + OFF_VE + 3932160);
    float* vbuf   = (float*)(ws + OFF_VE + 5898240);
    float* attb   = (float*)(ws + OFF_VE + 7864320);
    float* aobuf  = (float*)(ws + OFF_VE + 11550720);
    float* x1pre  = (float*)(ws + OFF_VE + 13516800);
    float* x1     = (float*)(ws + OFF_VE + 15482880);
    float* midb   = (float*)(ws + OFF_VE + 17448960);
    float* x2pre  = (float*)(ws + OFF_VE + 19415040);
    float* within = (float*)(ws + OFF_VE + 21381120);

    auto cdiv = [](int a, int b) { return (a + b - 1) / b; };

    // ---- init LSTM state ----
    zero_f<<<cdiv(4096, 256), 256, 0, stream>>>(hbuf, 4096);
    zero_f<<<cdiv(4096, 256), 256, 0, stream>>>(cbuf, 4096);

    // ---- weight swizzles (bf16 WMMA B-fragments) ----
    auto sw = [&](const float* W, __bf16* dst, int K, int N) {
        int threads = (K >> 5) * (N >> 4) * 32;
        swizzle_w<<<cdiv(threads, 256), 256, 0, stream>>>(W, dst, K, N);
    };
    sw(W_ev, wev_sw, 2048, 256);
    sw(W_sp1, w1a_sw, 256, 256);        // top 256 rows of W_sp1
    sw(W_tv, wtv_sw, 512, 512);
    sw(Wq, wq_sw, 512, 512);
    sw(Wk, wk_sw, 512, 512);
    sw(Wv, wv_sw, 512, 512);
    sw(Wo, wo_sw, 512, 512);
    sw(Wf1, wf1_sw, 512, 512);
    sw(Wf2, wf2_sw, 512, 512);
    sw(W_l1, wl1a_sw, 512, 512);        // top 512 rows of W_l1

    // ---- ve (dominant WMMA GEMM, 2400 waves, videos read once, async-LDS B) ----
    ve_kernel<<<600, 128, 0, stream>>>(videos, wev_sw, b_ev, W_el, b_el, ve);

    // ---- word embeddings + per-word bias terms for attend ----
    vecmat<<<cdiv(2048, 256), 256, 0, stream>>>(sub_glove, W_ew, b_ew, sub_e, cB, cWD, cE, 0, 1);
    vecmat<<<cdiv(2048, 256), 256, 0, stream>>>(obj_glove, W_ew, b_ew, obj_e, cB, cWD, cE, 0, 1);
    vecmat<<<cdiv(2048, 256), 256, 0, stream>>>(sub_e, W_sp1, b_sp1, wts, cB, cE, cE, cE, 0);
    vecmat<<<cdiv(2048, 256), 256, 0, stream>>>(obj_e, W_sp1, b_sp1, wto, cB, cE, cE, cE, 0);

    // ---- Y = ve @ W_sp1[:256] (shared by both attend calls) ----
    gemm_wmma<<<cdiv(2400 * 2, 4), 128, 0, stream>>>(ve, w1a_sw, nullptr, nullptr, Y,
                                                     cM, cE, cE, cE, cE, 0);
    attend_scores<<<cM, 256, 0, stream>>>(Y, wts, wto, w_sp2, s_sub, s_obj);
    softmax_nb<<<cdiv(cB * cT, 256), 256, 0, stream>>>(s_sub, s_obj, out_sub, out_obj);
    build_orix<<<cdiv(cB * cT * cH, 256), 256, 0, stream>>>(ve, out_sub, out_obj,
                                                            W_s2o, b_s2o, W_o2s, b_o2s, orix);

    // ---- transformer block ----
    gemm_wmma<<<60, 128, 0, stream>>>(orix, wtv_sw, b_tv, nullptr, xbuf, 960, 512, 512, 512, 512, 1);
    gemm_wmma<<<60, 128, 0, stream>>>(xbuf, wq_sw, bq, nullptr, qbuf, 960, 512, 512, 512, 512, 0);
    gemm_wmma<<<60, 128, 0, stream>>>(xbuf, wk_sw, bk, nullptr, kbuf, 960, 512, 512, 512, 512, 0);
    gemm_wmma<<<60, 128, 0, stream>>>(xbuf, wv_sw, bv, nullptr, vbuf, 960, 512, 512, 512, 512, 0);
    attn_scores<<<cB * cNH * cT, 128, 0, stream>>>(qbuf, kbuf, attb);
    attn_out<<<cdiv(cB * cT * cH, 256), 256, 0, stream>>>(attb, vbuf, aobuf);
    gemm_wmma<<<60, 128, 0, stream>>>(aobuf, wo_sw, bo, xbuf, x1pre, 960, 512, 512, 512, 512, 0);
    layernorm_k<<<960, 256, 0, stream>>>(x1pre, g1, be1, x1);
    gemm_wmma<<<60, 128, 0, stream>>>(x1, wf1_sw, bf1, nullptr, midb, 960, 512, 512, 512, 512, 1);
    gemm_wmma<<<60, 128, 0, stream>>>(midb, wf2_sw, bf2, x1, x2pre, 960, 512, 512, 512, 512, 0);
    layernorm_k<<<960, 256, 0, stream>>>(x2pre, g2, be2, within);

    // ---- segments, beta2 ----
    seg_gather<<<cdiv(cB * cNSEG * cH, 256), 256, 0, stream>>>(within, seg);
    vecmat2<<<cdiv(cB * cH, 256), 256, 0, stream>>>(sub_e, obj_e, W_tr, b_tr, trbuf,
                                                    cB, cE, cE, cH, 1);
    vecmat<<<cdiv(cB * cH, 256), 256, 0, stream>>>(trbuf, W_tb1, b_tb1, trterm, cB, cH, cH, cH, 0);
    beta2_scores<<<cB * cNSEG, 512, 0, stream>>>(seg, W_tb1, trterm, w_tb2, sc2);
    softmax10<<<1, 32, 0, stream>>>(sc2, out_b2);

    // ---- LSTM over NSEG steps ----
    for (int t = 0; t < cNSEG; t++) {
        lstm_gates<<<cdiv(cB * 4 * cH, 256), 256, 0, stream>>>(seg, hbuf, W_ih, W_hh,
                                                               b_ih, b_hh, gates, t);
        lstm_update<<<cdiv(cB * cH, 256), 256, 0, stream>>>(gates, hbuf, cbuf,
                                                            out_hT, out_cT, t == cNSEG - 1);
    }

    // ---- beta1 + final output ----
    vecmat<<<cdiv(cB * cH, 256), 256, 0, stream>>>(hbuf, W_l1, b_l1, hvec, cB, cH, cH, cH, 0);
    gemm_wmma<<<60, 128, 0, stream>>>(within, wl1a_sw, nullptr, nullptr, Ylw,
                                      960, 512, 512, 512, 512, 0);
    beta1_scores<<<cB * cT, 512, 0, stream>>>(Ylw, hvec, w_l2, s1);
    softmax_t<<<cB, 128, 0, stream>>>(s1, out_b1);
    final_out<<<cdiv(cB * cH, 256), 256, 0, stream>>>(within, out_b1, out_b2, out_output);
}
